// ConditionalHMM_86629490360843
// MI455X (gfx1250) — matrix operand order repaired and checked by reading, not compiled, not run on hardware
//
#include <hip/hip_runtime.h>
#include <hip/hip_bf16.h>

#define B_ 8
#define T_ 512
#define D_ 768
#define S_ 8
#define H_ 32
#define O_ 32
#define HH_ (H_*H_)   // 1024
#define HO_ (H_*O_)   // 1024
#define MROWS (B_*T_) // 4096
#define NEG_INF (-3.0e38f)

typedef __attribute__((ext_vector_type(8)))  __bf16 v8bf;
typedef __attribute__((ext_vector_type(16))) __bf16 v16bf;
typedef __attribute__((ext_vector_type(8)))  float  v8f;

// ---------------- wave32 helpers ----------------
__device__ __forceinline__ float wredMax(float v) {
    #pragma unroll
    for (int off = 16; off > 0; off >>= 1) v = fmaxf(v, __shfl_xor(v, off, 32));
    return v;
}
__device__ __forceinline__ float wredSum(float v) {
    #pragma unroll
    for (int off = 16; off > 0; off >>= 1) v += __shfl_xor(v, off, 32);
    return v;
}
__device__ __forceinline__ float logaddf(float a, float b) {
    float m = fmaxf(a, b);
    if (m <= NEG_INF) return NEG_INF;
    return m + logf(expf(a - m) + expf(b - m));
}

// ---------------- fp32 -> bf16 convert ----------------
__global__ void cvt_f32_bf16(const float* __restrict__ in, __bf16* __restrict__ out, size_t n) {
    size_t i = (size_t)blockIdx.x * blockDim.x + threadIdx.x;
    if (i < n) out[i] = (__bf16)in[i];
}

// ---------------- static param softmaxes ----------------
// blocks: 0..31 rows of unnorm_trans, 32..287 rows of unnorm_emiss, 288 = log_softmax(state_priors)
__global__ void params_prep(const float* __restrict__ ut, const float* __restrict__ ue,
                            const float* __restrict__ sp,
                            float* __restrict__ trans_sm, float* __restrict__ emiss_sm,
                            float* __restrict__ lsp) {
    int r = blockIdx.x;
    int lane = threadIdx.x;
    if (r < 32) {
        float x = ut[r * 32 + lane];
        float m = wredMax(x);
        float e = expf(x - m);
        float s = wredSum(e);
        trans_sm[r * 32 + lane] = e / s;
    } else if (r < 288) {
        int q = r - 32;
        float x = ue[q * 32 + lane];
        float m = wredMax(x);
        float e = expf(x - m);
        float s = wredSum(e);
        emiss_sm[q * 32 + lane] = e / s;
    } else {
        float x = sp[lane];
        float m = wredMax(x);
        float s = wredSum(expf(x - m));
        lsp[lane] = x - (m + logf(s));
    }
}

// ---------------- bf16 WMMA GEMM, register double-buffered ----------------
// C[row*cRowStride + z*cZStride + col] = sum_k A[row,k]*B[z][k,col] + bias[z][col]
// Block: 256 thr = 8 waves as 2(M) x 4(N); wave tile 16x64; block tile 32x256.
union ABfrag { v16bf v; v8bf h[2]; };

__global__ __launch_bounds__(256)
void gemm_bf16_wmma(const __bf16* __restrict__ A, const __bf16* __restrict__ B0,
                    const float* __restrict__ bias0, float* __restrict__ C,
                    int N, int K,
                    size_t bZStride, size_t biasZStride,
                    size_t cRowStride, size_t cZStride) {
    const int z = blockIdx.z;
    const __bf16* Bm  = B0 + (size_t)z * bZStride;
    const float* bias = bias0 + (size_t)z * biasZStride;

    const int wave = threadIdx.x >> 5;
    const int lane = threadIdx.x & 31;
    const int wm = wave >> 2;           // 0..1
    const int wn = wave & 3;            // 0..3
    const int m0 = blockIdx.y * 32 + wm * 16;
    const int n0 = blockIdx.x * 256 + wn * 64;

    const int g    = lane >> 4;         // half-wave group
    const int mrow = m0 + (lane & 15);  // A row held by this lane

    // running pointers (advance by one 32-wide K chunk each step)
    const __bf16* ap = A + (size_t)mrow * K + g * 8;     // A frag: [ap, ap+8) and [ap+16, ap+24)
    const __bf16* bp = Bm + (size_t)(g * 8) * N + n0 + (lane & 15);   // B frag column for this lane
    const size_t bStep = (size_t)32 * N;

    v8f acc[4] = {};
    ABfrag a0, a1, b0[4], b1[4];

    auto loadFrag = [&](ABfrag& aa, ABfrag (&bb)[4]) {
        aa.h[0] = *(const v8bf*)(ap);
        aa.h[1] = *(const v8bf*)(ap + 16);
        #pragma unroll
        for (int nt = 0; nt < 4; ++nt) {
            #pragma unroll
            for (int i = 0; i < 8; ++i) {
                bb[nt].v[i]     = bp[(size_t)i * N + nt * 16];
                bb[nt].v[8 + i] = bp[(size_t)(16 + i) * N + nt * 16];
            }
        }
        __builtin_prefetch((const void*)(bp + 2 * bStep), 0, 1);   // global_prefetch 2 chunks ahead
        ap += 32;
        bp += bStep;
    };

    const int chunks = K / 32;          // 24 for K=768
    loadFrag(a0, b0);

    for (int c = 0; c < chunks; c += 2) {
        if (c + 1 < chunks) loadFrag(a1, b1);
        #pragma unroll
        for (int nt = 0; nt < 4; ++nt)
            acc[nt] = __builtin_amdgcn_wmma_f32_16x16x32_bf16(
                false, a0.v, false, b0[nt].v, (short)0, acc[nt], false, false);
        if (c + 2 < chunks) loadFrag(a0, b0);
        if (c + 1 < chunks) {
            #pragma unroll
            for (int nt = 0; nt < 4; ++nt)
                acc[nt] = __builtin_amdgcn_wmma_f32_16x16x32_bf16(
                    false, a1.v, false, b1[nt].v, (short)0, acc[nt], false, false);
        }
    }

    // C/D 16x16 f32 layout: col = n-sub + lane%16, row = m0 + g*8 + vgpr index
    #pragma unroll
    for (int nt = 0; nt < 4; ++nt) {
        int col = n0 + nt * 16 + (lane & 15);
        float bv = bias[col];
        #pragma unroll
        for (int r = 0; r < 8; ++r) {
            int row = m0 + g * 8 + r;
            C[(size_t)row * cRowStride + (size_t)z * cZStride + col] = acc[nt][r] + bv;
        }
    }
}

// ---------------- in-place: log(0.5*base_sm + 0.5*softmax(logits)) ----------------
// one wave per 32-wide row; base row index = (row % baseMod)
__global__ __launch_bounds__(256)
void combine_log_kernel(float* __restrict__ logits, const float* __restrict__ base_sm,
                        int nRows, int baseMod) {
    int w = blockIdx.x * (blockDim.x >> 5) + (threadIdx.x >> 5);
    if (w >= nRows) return;
    int lane = threadIdx.x & 31;
    size_t idx = (size_t)w * 32 + lane;
    float x = logits[idx];
    float m = wredMax(x);
    float e = expf(x - m);
    float s = wredSum(e);
    float p = e / s;
    float bse = base_sm[(size_t)(w % baseMod) * 32 + lane];
    logits[idx] = logf(0.5f * bse + 0.5f * p);
}

// ---------------- normalize_observation -> log(obs') ----------------
// one wave per (b,t); lane = o
__global__ __launch_bounds__(256)
void logobs_kernel(const float* __restrict__ obs, float* __restrict__ lobs, int nBT) {
    int w = blockIdx.x * (blockDim.x >> 5) + (threadIdx.x >> 5);
    if (w >= nBT) return;
    int lane = threadIdx.x & 31;
    int lbs[S_];
    int ssum = 0;
    #pragma unroll
    for (int s = 0; s < S_; ++s) {
        float v = obs[(size_t)w * (S_ * O_) + s * O_ + lane];
        int idx = lane;
        #pragma unroll
        for (int off = 16; off > 0; off >>= 1) {
            float ov = __shfl_xor(v, off, 32);
            int oi = __shfl_xor(idx, off, 32);
            if (ov > v || (ov == v && oi < idx)) { v = ov; idx = oi; }
        }
        lbs[s] = idx;
        ssum += idx;
    }
    bool entity = (ssum > 0);
    #pragma unroll
    for (int s = 0; s < S_; ++s) {
        size_t q = (size_t)w * (S_ * O_) + s * O_ + lane;
        float ov = obs[q];
        bool mask = entity && (lbs[s] == 0);
        float val = mask ? (lane == 0 ? 0.01f : (0.99f / 32.0f)) : ov;
        lobs[q] = logf(val);
    }
}

// ---------------- lep[b,t,h] = sum_s LSE_o(log_emiss + log obs') ----------------
// one wave per (b,t,h); lane = o
__global__ __launch_bounds__(256)
void lep_kernel(const float* __restrict__ le, const float* __restrict__ lobs,
                float* __restrict__ lep, int nRows) {
    int w = blockIdx.x * (blockDim.x >> 5) + (threadIdx.x >> 5);
    if (w >= nRows) return;
    int lane = threadIdx.x & 31;
    int bt = w >> 5;
    int h  = w & 31;
    float acc = 0.0f;
    #pragma unroll
    for (int s = 0; s < S_; ++s) {
        float v = le[(size_t)bt * (S_ * HO_) + (size_t)s * HO_ + h * 32 + lane]
                + lobs[(size_t)bt * (S_ * O_) + s * O_ + lane];
        float m = wredMax(v);
        float sm = wredSum(expf(v - m));
        acc += m + logf(sm);
    }
    if (lane == 0) lep[w] = acc;
}

// ---------------- forward recursion: one wave per batch, lane = h ----------------
__global__ void forward_kernel(const float* __restrict__ lt, const float* __restrict__ lep,
                               const float* __restrict__ lsp, float* __restrict__ alpha) {
    int b = blockIdx.x;
    int lane = threadIdx.x;
    float a = lsp[lane] + lep[(size_t)b * T_ * H_ + lane];
    float m = wredMax(a);
    float s = wredSum(expf(a - m));
    a -= (m + logf(s));
    alpha[(size_t)b * T_ * H_ + lane] = a;
    for (int t = 1; t < T_; ++t) {
        const float* tr = lt + ((size_t)b * T_ + t) * HH_;
        float acc = NEG_INF;
        for (int hp = 0; hp < 32; ++hp) {
            float av = __shfl(a, hp, 32);
            acc = logaddf(acc, av + tr[hp * 32 + lane]);
        }
        float la = lep[((size_t)b * T_ + t) * H_ + lane] + acc;
        m = wredMax(la);
        s = wredSum(expf(la - m));
        a = la - (m + logf(s));
        alpha[((size_t)b * T_ + t) * H_ + lane] = a;
    }
}

// ---------------- backward recursion (pre-roll): lane = h' ----------------
__global__ void backward_kernel(const float* __restrict__ lt, const float* __restrict__ lep,
                                float* __restrict__ braw) {
    int b = blockIdx.x;
    int lane = threadIdx.x;
    float bv = 0.0f;
    braw[((size_t)b * T_ + (T_ - 1)) * H_ + lane] = 0.0f;
    for (int t = T_ - 2; t >= 0; --t) {
        float c = lep[((size_t)b * T_ + t) * H_ + lane] + bv;   // lane holds h = lane
        const float* tr = lt + ((size_t)b * T_ + t) * HH_;
        float acc = NEG_INF;
        for (int h = 0; h < 32; ++h) {
            float cv = __shfl(c, h, 32);
            acc = logaddf(acc, tr[lane * 32 + h] + cv);
        }
        bv = acc;
        braw[((size_t)b * T_ + t) * H_ + lane] = acc;
    }
}

// ---------------- roll beta by seq_len - T along time ----------------
__global__ void roll_kernel(const float* __restrict__ raw, float* __restrict__ outb,
                            const int* __restrict__ seq) {
    size_t i = (size_t)blockIdx.x * blockDim.x + threadIdx.x;
    if (i >= (size_t)B_ * T_ * H_) return;
    int h = i & 31;
    size_t bt = i >> 5;
    int t = (int)(bt % T_);
    int b = (int)(bt / T_);
    int ts = t + (T_ - seq[b]);
    if (ts >= T_) ts -= T_;
    outb[i] = raw[((size_t)b * T_ + ts) * H_ + h];
}

// ---------------- ll init ----------------
__global__ void zero_ll(float* ll) { if (threadIdx.x == 0 && blockIdx.x == 0) ll[0] = 0.0f; }

// ---------------- gamma: prior + emission expectation terms ----------------
// one wave per (b,t); lane = h
__global__ __launch_bounds__(256)
void gamma_kernel(const float* __restrict__ alpha, const float* __restrict__ beta,
                  const float* __restrict__ lep, const float* __restrict__ lsp,
                  const int* __restrict__ seq, float* __restrict__ ll) {
    int w = blockIdx.x * (blockDim.x >> 5) + (threadIdx.x >> 5);
    if (w >= B_ * T_) return;
    int lane = threadIdx.x & 31;
    int b = w / T_, t = w % T_;
    float g = alpha[(size_t)w * H_ + lane] + beta[(size_t)w * H_ + lane];
    float m = wredMax(g);
    float s = wredSum(expf(g - m));
    float ng = g - (m + logf(s));
    float p = expf(ng);
    float contrib = 0.0f;
    float et = wredSum(p * lep[(size_t)w * H_ + lane]);
    if (t < seq[b]) contrib += et;
    float pt = wredSum(p * lsp[lane]);
    if (t == 0) contrib += pt;
    if (lane == 0) atomicAdd(ll, contrib * (1.0f / B_));
}

// ---------------- xi: transition expectation term (online normalized) ----------------
// one wave per (b, t-1); lane covers h, loop covers h'; never materializes xi
__global__ __launch_bounds__(256)
void xi_kernel(const float* __restrict__ lt, const float* __restrict__ alpha,
               const float* __restrict__ beta, const float* __restrict__ lep,
               const int* __restrict__ seq, float* __restrict__ ll) {
    int w = blockIdx.x * (blockDim.x >> 5) + (threadIdx.x >> 5);
    if (w >= B_ * (T_ - 1)) return;
    int lane = threadIdx.x & 31;
    int b = w / (T_ - 1), tm1 = w % (T_ - 1), t = tm1 + 1;
    const float* tr = lt + ((size_t)b * T_ + t) * HH_;
    float alv = alpha[((size_t)b * T_ + tm1) * H_ + lane];
    float wl  = lep[((size_t)b * T_ + t) * H_ + lane]
              + beta[((size_t)b * T_ + t) * H_ + lane];
    float m = NEG_INF, se = 0.0f, sw = 0.0f;
    for (int k = 0; k < 32; ++k) {
        float av  = __shfl(alv, k, 32);
        float trv = tr[k * 32 + lane];
        float v = trv + av + wl;
        if (v > m) { float sc = expf(m - v); se *= sc; sw *= sc; m = v; }
        float e = expf(v - m);
        se += e;
        sw += e * trv;
    }
    #pragma unroll
    for (int off = 16; off > 0; off >>= 1) {
        float om  = __shfl_xor(m, off, 32);
        float ose = __shfl_xor(se, off, 32);
        float osw = __shfl_xor(sw, off, 32);
        if (om > m) { float sc = expf(m - om); se = se * sc + ose; sw = sw * sc + osw; m = om; }
        else        { float sc = expf(om - m); se += ose * sc;     sw += osw * sc; }
    }
    float term = sw / se;
    if (lane == 0 && tm1 < seq[b] - 1) atomicAdd(ll, term * (1.0f / B_));
}

// ================= host launcher =================
extern "C" void kernel_launch(void* const* d_in, const int* in_sizes, int n_in,
                              void* d_out, int out_size, void* d_ws, size_t ws_size,
                              hipStream_t stream) {
    const float* emb = (const float*)d_in[0];
    const float* obs = (const float*)d_in[1];
    const int*   seq = (const int*)d_in[2];
    const float* sp  = (const float*)d_in[3];
    const float* ut  = (const float*)d_in[4];
    const float* ue  = (const float*)d_in[5];
    const float* Wt  = (const float*)d_in[6];
    const float* bt  = (const float*)d_in[7];
    const float* We  = (const float*)d_in[8];
    const float* be  = (const float*)d_in[9];

    float* outf = (float*)d_out;
    float* d_ll = outf;                                    // [1]
    float* d_lt = outf + 1;                                // [B,T,H,H]
    float* d_le = d_lt + (size_t)B_ * T_ * HH_;            // [B,T,S,H,O]

    // workspace carve-up
    char* w = (char*)d_ws;
    size_t off = 0;
    auto alloc = [&](size_t bytes) -> void* {
        void* p = w + off;
        off = (off + bytes + 255) & ~(size_t)255;
        return p;
    };
    __bf16* embB = (__bf16*)alloc((size_t)MROWS * D_ * sizeof(__bf16));
    __bf16* WtB  = (__bf16*)alloc((size_t)D_ * HH_ * sizeof(__bf16));
    __bf16* WeB  = (__bf16*)alloc((size_t)S_ * D_ * HO_ * sizeof(__bf16));
    float* trans_sm = (float*)alloc((size_t)HH_ * sizeof(float));
    float* emiss_sm = (float*)alloc((size_t)S_ * HO_ * sizeof(float));
    float* lspw     = (float*)alloc((size_t)H_ * sizeof(float));
    float* lobs  = (float*)alloc((size_t)B_ * T_ * S_ * O_ * sizeof(float));
    float* lepw  = (float*)alloc((size_t)B_ * T_ * H_ * sizeof(float));
    float* alpha = (float*)alloc((size_t)B_ * T_ * H_ * sizeof(float));
    float* braw  = (float*)alloc((size_t)B_ * T_ * H_ * sizeof(float));
    float* beta  = (float*)alloc((size_t)B_ * T_ * H_ * sizeof(float));

    // 1) fp32 -> bf16
    {
        size_t n1 = (size_t)MROWS * D_;
        cvt_f32_bf16<<<(unsigned)((n1 + 255) / 256), 256, 0, stream>>>(emb, embB, n1);
        size_t n2 = (size_t)D_ * HH_;
        cvt_f32_bf16<<<(unsigned)((n2 + 255) / 256), 256, 0, stream>>>(Wt, WtB, n2);
        size_t n3 = (size_t)S_ * D_ * HO_;
        cvt_f32_bf16<<<(unsigned)((n3 + 255) / 256), 256, 0, stream>>>(We, WeB, n3);
    }

    // 2) static param softmaxes
    params_prep<<<289, 32, 0, stream>>>(ut, ue, sp, trans_sm, emiss_sm, lspw);

    // 3) WMMA GEMMs (raw logits + bias straight into output regions)
    {
        dim3 grid(HH_ / 256, MROWS / 32, 1);
        gemm_bf16_wmma<<<grid, 256, 0, stream>>>(embB, WtB, bt, d_lt,
                                                 HH_, D_, 0, 0,
                                                 (size_t)HH_, 0);
        dim3 gride(HO_ / 256, MROWS / 32, S_);
        gemm_bf16_wmma<<<gride, 256, 0, stream>>>(embB, WeB, be, d_le,
                                                  HO_, D_,
                                                  (size_t)D_ * HO_, (size_t)HO_,
                                                  (size_t)S_ * HO_, (size_t)HO_);
    }

    // 4) in-place softmax + mix + log
    {
        int rowsT = MROWS * H_;                       // 131072
        combine_log_kernel<<<rowsT / 8, 256, 0, stream>>>(d_lt, trans_sm, rowsT, H_);
        int rowsE = MROWS * S_ * H_;                  // 1048576
        combine_log_kernel<<<rowsE / 8, 256, 0, stream>>>(d_le, emiss_sm, rowsE, S_ * H_);
    }

    // 5) normalized log-observations, then lep
    logobs_kernel<<<MROWS / 8, 256, 0, stream>>>(obs, lobs, MROWS);
    {
        int rows = MROWS * H_;
        lep_kernel<<<rows / 8, 256, 0, stream>>>(d_le, lobs, lepw, rows);
    }

    // 6) HMM recursions
    forward_kernel<<<B_, 32, 0, stream>>>(d_lt, lepw, lspw, alpha);
    backward_kernel<<<B_, 32, 0, stream>>>(d_lt, lepw, braw);
    {
        size_t n = (size_t)B_ * T_ * H_;
        roll_kernel<<<(unsigned)((n + 255) / 256), 256, 0, stream>>>(braw, beta, seq);
    }

    // 7) log-likelihood terms
    zero_ll<<<1, 32, 0, stream>>>(d_ll);
    gamma_kernel<<<(B_ * T_) / 8, 256, 0, stream>>>(alpha, beta, lepw, lspw, seq, d_ll);
    {
        int nw = B_ * (T_ - 1);                       // 4088, divisible by 8
        xi_kernel<<<nw / 8, 256, 0, stream>>>(d_lt, alpha, beta, lepw, seq, d_ll);
    }
}
